// EarthSpecificLayerAbsolute_16595753631902
// MI455X (gfx1250) — matrix operand order repaired and checked by reading, not compiled, not run on hardware
//
#include <hip/hip_runtime.h>
#include <hip/hip_bf16.h>

typedef __attribute__((ext_vector_type(16))) _Float16 v16h;
typedef __attribute__((ext_vector_type(8)))  float    v8f;

#define NWIN   960
#define PN     144
#define DIM    192
#define HEADS  6
#define DHEAD  32
#define TWIN   64
#define PQ     (PN * PN)          /* 20736 */
#define QKSCALE 0.17677669529663687f  /* 32^-0.5 */

// D = A(16x32 f16) * B(32x16 f16) + C(16x16 f32)
static __device__ __forceinline__ v8f wmma16(v16h a, v16h b, v8f c) {
  return __builtin_amdgcn_wmma_f32_16x16x32_f16(false, a, false, b, (short)0, c,
                                                false, false);
}

// ---------------------------------------------------------------------------
// Kernel 0: convert fp32 weights [K][N] into f16 "B-fragment order":
//   Wb[ct][kc][half][n15][k15]  where half = which 16-lane group (K 0-15/16-31
//   of the chunk).  A lane's whole B fragment is then 32 contiguous bytes.
// ---------------------------------------------------------------------------
__global__ void wconv_kernel(const float* __restrict__ W,
                             _Float16* __restrict__ Wb, int K, int N) {
  int g = blockIdx.x * 256 + threadIdx.x;
  if (g >= K * N) return;
  const int k = g / N, n = g % N;
  const int ct = n >> 4, nn = n & 15;
  const int kc = k >> 5, half = (k >> 4) & 1, ki = k & 15;
  const int KC = K >> 5;
  Wb[((((ct * KC + kc) * 2 + half) * 16 + nn) * 16) + ki] = (_Float16)W[g];
}

// ---------------------------------------------------------------------------
// Kernel 1: QKV projection.  qkv = x @ W_qkv + b_qkv ; split into per-head
// f16 Q (pre-scaled), K, V in workspace, layout [w][h][p][d].
// One wave per 16x16 output tile, K=192 in 6 chunks of 32.
// ---------------------------------------------------------------------------
__global__ void __launch_bounds__(256) qkv_kernel(
    const float* __restrict__ x, const _Float16* __restrict__ Wb,
    const float* __restrict__ bq,
    _Float16* __restrict__ qws, _Float16* __restrict__ kws,
    _Float16* __restrict__ vws) {
  const int lane = threadIdx.x & 31;
  const int wid  = threadIdx.x >> 5;
  const int NT_COL = (3 * DIM) / 16;            // 36
  const int NT_ROW = (NWIN * PN) / 16;          // 8640
  int tile = blockIdx.x * 8 + wid;
  if (tile >= NT_ROW * NT_COL) return;
  const int rt = tile / NT_COL, ct = tile % NT_COL;
  const int col = ct * 16 + (lane & 15);

  const float bv = bq[col];
  v8f c;
#pragma unroll
  for (int v = 0; v < 8; ++v) c[v] = bv;

  const int arow = rt * 16 + (lane & 15);
  const int kb_a = (lane < 16) ? 0 : 16;        // A-frag K base (contiguous layout)
  const int bhalf = (lane < 16) ? 0 : 1;
  // per-lane base into fragment-ordered weights for this column tile
  const _Float16* wb0 =
      Wb + ((((size_t)ct * 6) * 2 + bhalf) * 16 + (lane & 15)) * 16;

#pragma unroll
  for (int kc = 0; kc < 6; ++kc) {
    const int k0 = kc * 32;
    v16h a, b;
    const float* ar = x + (size_t)arow * DIM + k0 + kb_a;
#pragma unroll
    for (int v = 0; v < 8; ++v) {               // pairs at {0,2,4,...,14}
      const int kk = 2 * v;
      a[2 * v]     = (_Float16)ar[kk];
      a[2 * v + 1] = (_Float16)ar[kk + 1];
    }
    const _Float16* wc = wb0 + (size_t)kc * (2 * 16 * 16);
    if (kc < 5) __builtin_prefetch(wc + 2 * 16 * 16, 0, 3);
#pragma unroll
    for (int i = 0; i < 16; ++i) b[i] = wc[i];  // 32 contiguous bytes
    c = wmma16(a, b, c);
  }

  // epilogue: scatter into Q/K/V per-head layout
  const int which = col / DIM;                  // 0=q 1=k 2=v (tile never crosses)
  const int rem   = col % DIM;
  const int h = rem / DHEAD, d = rem % DHEAD;
  _Float16* dst = (which == 0) ? qws : ((which == 1) ? kws : vws);
  const float mul = (which == 0) ? QKSCALE : 1.0f;
  const int m0 = (lane >> 4) << 3;
#pragma unroll
  for (int v = 0; v < 8; ++v) {
    const int grow = rt * 16 + m0 + v;          // global row (may cross windows)
    const int w = grow / PN, p = grow % PN;
    dst[(((size_t)w * HEADS + h) * PN + p) * DHEAD + d] = (_Float16)(c[v] * mul);
  }
}

// ---------------------------------------------------------------------------
// Kernel 2: expand gathered position bias -> bias_full[t][h][p][q] (31.8 MB,
// L2-resident; each (t,h) plane is reused by 15 windows).
// ---------------------------------------------------------------------------
__global__ void bias_expand(const float* __restrict__ btab,
                            const int* __restrict__ pidx,
                            float* __restrict__ bf) {
  int g = blockIdx.x * 256 + threadIdx.x;
  if (g >= TWIN * HEADS * PQ) return;
  const int th = g / PQ;                        // t*6+h
  const int pq = g % PQ;
  bf[g] = btab[(size_t)pidx[pq] * (TWIN * HEADS) + th];
}

// ---------------------------------------------------------------------------
// Kernel 3: attention for one (window, head) per block.  8 waves / block.
// LDS: S f32 [144][160] (V staging first, f16 probs [144][320] later — both
//      alias S), Q f16 [144][32], K f16 [144][32], Vt f16 [32][160].
// Q/K/V staged with GLOBAL_LOAD_ASYNC_TO_LDS_B128 (ASYNCcnt path).
// ---------------------------------------------------------------------------
__global__ void __launch_bounds__(256) attn_kernel(
    const _Float16* __restrict__ qws, const _Float16* __restrict__ kws,
    const _Float16* __restrict__ vws, const float* __restrict__ biasf,
    const float* __restrict__ mask, _Float16* __restrict__ obuf) {
  extern __shared__ char smem[];
  float*    S  = (float*)smem;                  // [144][160] scores
  _Float16* Ph = (_Float16*)smem;               // alias: probs, row stride 320
  _Float16* Vst = (_Float16*)smem;              // alias: V staging [144][32]
  _Float16* Qs = (_Float16*)(smem + PN * 160 * 4);
  _Float16* Ks = Qs + PN * DHEAD;
  _Float16* Vt = Ks + PN * DHEAD;               // [32][160] transposed V

  const int tid = threadIdx.x;
  const int lane = tid & 31, wid = tid >> 5;
  const int wh = blockIdx.x;                    // 0..5759
  const int w = wh / HEADS, h = wh % HEADS;
  const int t = w % TWIN;
  const size_t base = (size_t)wh * PN * DHEAD;  // wh == w*HEADS + h

  // ---- async DMA stage: Q, K -> LDS; V -> linear staging (in S region)
  {
    const _Float16* qg = qws + base;
    const _Float16* kg = kws + base;
    const _Float16* vg = vws + base;
    for (int i = tid; i < (PN * DHEAD) / 8; i += 256) {   // 576 x 16B each
      const int e = i * 8;
      unsigned lq = (unsigned)(size_t)(void*)(Qs + e);
      unsigned lk = (unsigned)(size_t)(void*)(Ks + e);
      unsigned lv = (unsigned)(size_t)(void*)(Vst + e);
      unsigned long long gq = (unsigned long long)(size_t)(qg + e);
      unsigned long long gk = (unsigned long long)(size_t)(kg + e);
      unsigned long long gv = (unsigned long long)(size_t)(vg + e);
      asm volatile("global_load_async_to_lds_b128 %0, %1, off"
                   :: "v"(lq), "v"(gq) : "memory");
      asm volatile("global_load_async_to_lds_b128 %0, %1, off"
                   :: "v"(lk), "v"(gk) : "memory");
      asm volatile("global_load_async_to_lds_b128 %0, %1, off"
                   :: "v"(lv), "v"(gv) : "memory");
    }
    asm volatile("s_wait_asynccnt 0x0" ::: "memory");
  }
  __syncthreads();

  // ---- transpose V (staging -> Vt), zero pad cols 144..159
  for (int i = tid; i < PN * DHEAD; i += 256) {
    const int p = i >> 5, d = i & 31;
    Vt[d * 160 + p] = Vst[i];
  }
  for (int i = tid; i < DHEAD * 16; i += 256) {
    Vt[(i >> 4) * 160 + PN + (i & 15)] = (_Float16)0.f;
  }
  __syncthreads();

  const float* bias_wh = biasf + (size_t)(t * HEADS + h) * PQ;
  const float* mask_w  = mask + (size_t)w * PQ;
  const int kb_a = (lane < 16) ? 0 : 16;
  const int kb_b = (lane < 16) ? 0 : 16;
  const int n  = lane & 15;
  const int m0 = (lane >> 4) << 3;

  // ---- scores: 81 tiles of 16x16, each ONE wmma (K = D = 32) + bias + mask
  for (int tI = wid; tI < 81; tI += 8) {
    const int pt = tI / 9, qt = tI % 9;
    v16h a, b;
    const _Float16* qr = Qs + (pt * 16 + (lane & 15)) * DHEAD + kb_a;
#pragma unroll
    for (int v = 0; v < 8; ++v) {
      const int kk = 2 * v;
      a[2 * v] = qr[kk]; a[2 * v + 1] = qr[kk + 1];
    }
    const _Float16* kr = Ks + (qt * 16 + (lane & 15)) * DHEAD + kb_b;
#pragma unroll
    for (int i = 0; i < 16; ++i) b[i] = kr[i];   // contiguous 32B: B = K^T
    v8f c = {};
    c = wmma16(a, b, c);
#pragma unroll
    for (int v = 0; v < 8; ++v) {
      const int p = pt * 16 + m0 + v, q = qt * 16 + n;
      S[p * 160 + q] = c[v] + bias_wh[p * PN + q] + mask_w[p * PN + q];
    }
  }
  __syncthreads();

  // ---- softmax, 18 rows per wave; write f16 probs in place (alias S)
  for (int r = wid * 18; r < wid * 18 + 18; ++r) {
    float vals[5];
    float mx = -3.0e38f;
#pragma unroll
    for (int k = 0; k < 5; ++k) {
      const int j = lane + 32 * k;
      vals[k] = (j < PN) ? S[r * 160 + j] : -3.0e38f;
      mx = fmaxf(mx, vals[k]);
    }
#pragma unroll
    for (int o = 16; o > 0; o >>= 1) mx = fmaxf(mx, __shfl_xor(mx, o, 32));
    float sum = 0.f;
#pragma unroll
    for (int k = 0; k < 5; ++k) {
      const int j = lane + 32 * k;
      vals[k] = (j < PN) ? __expf(vals[k] - mx) : 0.f;
      sum += vals[k];
    }
#pragma unroll
    for (int o = 16; o > 0; o >>= 1) sum += __shfl_xor(sum, o, 32);
    const float inv = 1.0f / sum;
#pragma unroll
    for (int k = 0; k < 5; ++k) {               // all reads done; safe to alias
      const int j = lane + 32 * k;
      Ph[r * 320 + j] = (_Float16)(vals[k] * inv);
    }
  }
  __syncthreads();

  // ---- O = P @ V : 9x2 tiles, K=160 in 5 chunks (cols 144..159 are zero)
  for (int tI = wid; tI < 18; tI += 8) {
    const int pt = tI >> 1, dt = tI & 1;
    v8f c = {};
    const int arow = pt * 16 + (lane & 15);
    const int vrow = dt * 16 + (lane & 15);
#pragma unroll
    for (int kc = 0; kc < 5; ++kc) {
      const int k0 = kc * 32;
      v16h a, b;
      const _Float16* pr = Ph + arow * 320 + k0 + kb_a;
#pragma unroll
      for (int v = 0; v < 8; ++v) {
        const int kk = 2 * v;
        a[2 * v] = pr[kk]; a[2 * v + 1] = pr[kk + 1];
      }
      const _Float16* vr = Vt + vrow * 160 + k0 + kb_b;  // contiguous (V^T)
#pragma unroll
      for (int i = 0; i < 16; ++i) b[i] = vr[i];
      c = wmma16(a, b, c);
    }
    const int col = h * DHEAD + dt * 16 + n;
#pragma unroll
    for (int v = 0; v < 8; ++v) {
      const int p = pt * 16 + m0 + v;
      obuf[((size_t)w * PN + p) * DIM + col] = (_Float16)c[v];
    }
  }
}

// ---------------------------------------------------------------------------
// Kernel 4: output projection  out = O @ W_out + b_out  (fp32 out)
// ---------------------------------------------------------------------------
__global__ void __launch_bounds__(256) outproj_kernel(
    const _Float16* __restrict__ obuf, const _Float16* __restrict__ Wb,
    const float* __restrict__ bo, float* __restrict__ out) {
  const int lane = threadIdx.x & 31;
  const int wid  = threadIdx.x >> 5;
  const int NT_COL = DIM / 16;                  // 12
  const int NT_ROW = (NWIN * PN) / 16;          // 8640
  int tile = blockIdx.x * 8 + wid;
  if (tile >= NT_ROW * NT_COL) return;
  const int rt = tile / NT_COL, ct = tile % NT_COL;
  const int col = ct * 16 + (lane & 15);

  const float bv = bo[col];
  v8f c;
#pragma unroll
  for (int v = 0; v < 8; ++v) c[v] = bv;

  const int arow = rt * 16 + (lane & 15);
  const int kb_a = (lane < 16) ? 0 : 16;
  const int bhalf = (lane < 16) ? 0 : 1;
  const _Float16* wb0 =
      Wb + ((((size_t)ct * 6) * 2 + bhalf) * 16 + (lane & 15)) * 16;
#pragma unroll
  for (int kc = 0; kc < 6; ++kc) {
    const int k0 = kc * 32;
    v16h a, b;
    const _Float16* ar = obuf + (size_t)arow * DIM + k0 + kb_a;
#pragma unroll
    for (int v = 0; v < 8; ++v) {
      const int kk = 2 * v;
      a[2 * v] = ar[kk]; a[2 * v + 1] = ar[kk + 1];
    }
    const _Float16* wc = wb0 + (size_t)kc * (2 * 16 * 16);
    if (kc < 5) __builtin_prefetch(wc + 2 * 16 * 16, 0, 3);
#pragma unroll
    for (int i = 0; i < 16; ++i) b[i] = wc[i];
    c = wmma16(a, b, c);
  }
  const int m0 = (lane >> 4) << 3;
#pragma unroll
  for (int v = 0; v < 8; ++v)
    out[(size_t)(rt * 16 + m0 + v) * DIM + col] = c[v];
}

// ---------------------------------------------------------------------------
extern "C" void kernel_launch(void* const* d_in, const int* in_sizes, int n_in,
                              void* d_out, int out_size, void* d_ws,
                              size_t ws_size, hipStream_t stream) {
  (void)in_sizes; (void)n_in; (void)out_size; (void)ws_size;
  const float* x    = (const float*)d_in[0];
  const float* mask = (const float*)d_in[1];
  const float* Wq   = (const float*)d_in[2];
  const float* bq   = (const float*)d_in[3];
  const float* Wo   = (const float*)d_in[4];
  const float* bo   = (const float*)d_in[5];
  const float* btab = (const float*)d_in[6];
  const int*   pidx = (const int*)d_in[7];
  float* out = (float*)d_out;

  char* ws = (char*)d_ws;
  const size_t qkvb = (size_t)NWIN * HEADS * PN * DHEAD * sizeof(_Float16);
  _Float16* qws   = (_Float16*)ws;  ws += qkvb;
  _Float16* kws   = (_Float16*)ws;  ws += qkvb;
  _Float16* vws   = (_Float16*)ws;  ws += qkvb;
  float*    bfull = (float*)ws;     ws += (size_t)TWIN * HEADS * PQ * sizeof(float);
  _Float16* obuf  = (_Float16*)ws;  ws += (size_t)NWIN * PN * DIM * sizeof(_Float16);
  _Float16* wbq   = (_Float16*)ws;  ws += (size_t)DIM * 3 * DIM * sizeof(_Float16);
  _Float16* wbo   = (_Float16*)ws;  ws += (size_t)DIM * DIM * sizeof(_Float16);

  {
    const int nq = DIM * 3 * DIM;                             // 110592
    wconv_kernel<<<(nq + 255) / 256, 256, 0, stream>>>(Wq, wbq, DIM, 3 * DIM);
    const int no = DIM * DIM;                                 // 36864
    wconv_kernel<<<(no + 255) / 256, 256, 0, stream>>>(Wo, wbo, DIM, DIM);
  }
  {
    const int tiles = ((NWIN * PN) / 16) * ((3 * DIM) / 16);  // 311040
    qkv_kernel<<<(tiles + 7) / 8, 256, 0, stream>>>(x, wbq, bq, qws, kws, vws);
  }
  {
    const int nel = TWIN * HEADS * PQ;                        // 7.96M
    bias_expand<<<(nel + 255) / 256, 256, 0, stream>>>(btab, pidx, bfull);
  }
  {
    const size_t shmem = (size_t)PN * 160 * 4                 // S / P / V-staging
                       + 2 * (size_t)PN * DHEAD * 2           // Q, K
                       + (size_t)DHEAD * 160 * 2;             // V^T  => 120832 B
    attn_kernel<<<NWIN * HEADS, 256, shmem, stream>>>(qws, kws, vws, bfull,
                                                      mask, obuf);
  }
  {
    const int tiles = ((NWIN * PN) / 16) * (DIM / 16);        // 103680
    outproj_kernel<<<(tiles + 7) / 8, 256, 0, stream>>>(obuf, wbo, bo, out);
  }
}
